// I2C_KNN_13907104104701
// MI455X (gfx1250) — matrix operand order repaired and checked by reading, not compiled, not run on hardware
//
#include <hip/hip_runtime.h>
#include <math.h>

typedef __attribute__((ext_vector_type(16))) _Float16 v16h;
typedef __attribute__((ext_vector_type(8)))  float    v8f;

#define HW       441
#define CCH      64
#define BQ       32
#define NSUP     25
#define CLASSES  5
#define POOL     2205            // per-class support descriptors (5*441)
#define STILES   138             // ceil(2205/16)
#define MQ       (BQ*HW)         // 14112 query descriptors
#define QTILES   882             // MQ/16 (exact)
#define NEG_BIG  (-1e30f)
#define EPSV     1e-8f

// ---------------------------------------------------------------------------
// Kernel 1: reciprocal L2 norms for all descriptors + zero the accumulators
// ---------------------------------------------------------------------------
__global__ void norm_init(const float* __restrict__ q, const float* __restrict__ S,
                          float* __restrict__ rnq, float* __restrict__ rns,
                          float* __restrict__ acc) {
  int i = blockIdx.x * blockDim.x + threadIdx.x;
  if (i < BQ * CLASSES) acc[i] = 0.0f;
  if (i < MQ) {
    int b = i / HW, hw = i % HW;
    float s = 0.0f;
    #pragma unroll 8
    for (int c = 0; c < CCH; ++c) {
      float x = q[((size_t)b * CCH + c) * HW + hw];
      s += x * x;
    }
    rnq[i] = 1.0f / sqrtf(s);
  } else if (i < MQ + NSUP * HW) {
    int j = i - MQ;
    int ns = j / HW, hw = j % HW;
    float s = 0.0f;
    #pragma unroll 8
    for (int c = 0; c < CCH; ++c) {
      float x = S[((size_t)ns * CCH + c) * HW + hw];
      s += x * x;
    }
    rns[j] = 1.0f / sqrtf(s);
  }
}

// ---------------------------------------------------------------------------
// Kernel 2: pack query into B-matrix (32x16 f16) lane fragments.
// B layout: lane L holds N = L&15; half h maps to K = 16*(L>>4) + h.
// Fragment index: ((qtile*2 + kf)*32 + lane) * 16 halfs (32B per lane).
// ---------------------------------------------------------------------------
__global__ void pack_b(const float* __restrict__ q, const float* __restrict__ rnq,
                       _Float16* __restrict__ bfrag) {
  int i = blockIdx.x * blockDim.x + threadIdx.x;
  if (i >= QTILES * 2 * 32) return;
  int lane = i & 31;
  int kf   = (i >> 5) & 1;
  int qt   = i >> 6;
  int hi   = lane >> 4;
  int n    = qt * 16 + (lane & 15);        // global query row (< 14112 always)
  int b = n / HW, hw = n % HW;
  float r = rnq[n];
  _Float16* dst = bfrag + (size_t)i * 16;
  #pragma unroll
  for (int h = 0; h < 16; ++h) {
    int k = kf * 32 + hi * 16 + h;
    dst[h] = (_Float16)(q[((size_t)b * CCH + k) * HW + hw] * r);
  }
}

// ---------------------------------------------------------------------------
// Kernel 3: pack support into A-matrix (16x32 f16) lane fragments.
// A layout: lane L holds M = L&15; half h maps to
//   K = 16*hi + h   (contiguous, matching the B packing).
// Fragment index: (((cls*STILES + st)*2 + kf)*32 + lane) * 16 halfs.
// Rows past the 2205-pool are zero-filled (masked again at top-k time).
// ---------------------------------------------------------------------------
__global__ void pack_a(const float* __restrict__ S, const float* __restrict__ rns,
                       _Float16* __restrict__ afrag) {
  int i = blockIdx.x * blockDim.x + threadIdx.x;
  if (i >= CLASSES * STILES * 2 * 32) return;
  int lane = i & 31;
  int kf   = (i >> 5) & 1;
  int tile = i >> 6;                 // cls*STILES + st
  int cls  = tile / STILES;
  int st   = tile % STILES;
  int hi   = lane >> 4;
  int m    = lane & 15;
  int jloc = st * 16 + m;
  _Float16* dst = afrag + (size_t)i * 16;
  if (jloc >= POOL) {
    #pragma unroll
    for (int h = 0; h < 16; ++h) dst[h] = (_Float16)0.0f;
    return;
  }
  int j  = cls * POOL + jloc;        // flat support descriptor index
  int ns = j / HW, hw = j % HW;
  float r = rns[j];
  #pragma unroll
  for (int h = 0; h < 16; ++h) {
    int kk = hi * 16 + h;
    int k  = kf * 32 + kk;
    dst[h] = (_Float16)(S[((size_t)ns * CCH + k) * HW + hw] * r);
  }
}

// swap data between lane L and L^16 (group-of-32 ds_swizzle, SWAPX16 = 0x401F)
__device__ inline float swap16(float x) {
  return __int_as_float(__builtin_amdgcn_ds_swizzle(__float_as_int(x), 0x401F));
}

// branchless sorted-top3 insert (t0 >= t1 >= t2)
__device__ inline void top3_insert(float x, float& t0, float& t1, float& t2) {
  float n0 = fmaxf(t0, x);
  float l0 = fminf(t0, x);
  float n1 = fmaxf(t1, l0);
  float l1 = fminf(t1, l0);
  float n2 = fmaxf(t2, l1);
  t0 = n0; t1 = n1; t2 = n2;
}

// ---------------------------------------------------------------------------
// Kernel 4: fused WMMA similarity + per-row top-3 over the class pool.
// One wave per (query 16-row tile, class). D tile: lane holds query col
// n = lane&15 and 8 support rows m = v + 8*(lane>>4) -> per-lane register
// top-3. Software-pipelined: while WMMA computes tile st, the top-3 VALU
// network consumes tile st-1 (fills the WMMA->VALU co-exec slots instead of
// v_nops). Only the peeled final tile applies the pool-boundary mask.
// ---------------------------------------------------------------------------
__global__ __launch_bounds__(128)
void knn_wmma(const _Float16* __restrict__ afrag, const _Float16* __restrict__ bfrag,
              float* __restrict__ acc) {
  const int lane = threadIdx.x & 31;
  const int wid  = threadIdx.x >> 5;
  const int job  = blockIdx.x * 4 + wid;
  if (job >= QTILES * CLASSES) return;
  const int cls   = job / QTILES;
  const int qtile = job % QTILES;
  const int hi    = lane >> 4;

  const v16h* aF = (const v16h*)afrag;
  const v16h* bF = (const v16h*)bfrag;

  // B fragments (this query tile) are loop-invariant: keep in registers
  const v16h b0 = bF[(size_t)(qtile * 2 + 0) * 32 + lane];
  const v16h b1 = bF[(size_t)(qtile * 2 + 1) * 32 + lane];

  float t0 = NEG_BIG, t1 = NEG_BIG, t2 = NEG_BIG;   // sorted desc
  const size_t abase = (size_t)cls * STILES * 2 * 32 + lane;

  // ---- prologue: tile 0
  v8f cprev = {};
  {
    const v16h a0 = aF[abase];
    const v16h a1 = aF[abase + 32];
    __builtin_prefetch((const void*)&aF[abase + 64], 0, 1);
    __builtin_prefetch((const void*)&aF[abase + 96], 0, 1);
    cprev = __builtin_amdgcn_wmma_f32_16x16x32_f16(false, a0, false, b0,
                                                   (short)0, cprev, false, false);
    cprev = __builtin_amdgcn_wmma_f32_16x16x32_f16(false, a1, false, b1,
                                                   (short)0, cprev, false, false);
  }

  // ---- steady state: WMMA(st) overlaps top-3 update of tile st-1
  for (int st = 1; st < STILES; ++st) {
    const size_t idx = abase + (size_t)st * 64;
    const v16h a0 = aF[idx];
    const v16h a1 = aF[idx + 32];
    if (st + 1 < STILES) {               // global_prefetch next A tile
      __builtin_prefetch((const void*)&aF[idx + 64], 0, 1);
      __builtin_prefetch((const void*)&aF[idx + 96], 0, 1);
    }
    v8f ccur = {};
    ccur = __builtin_amdgcn_wmma_f32_16x16x32_f16(false, a0, false, b0,
                                                  (short)0, ccur, false, false);
    ccur = __builtin_amdgcn_wmma_f32_16x16x32_f16(false, a1, false, b1,
                                                  (short)0, ccur, false, false);
    // tiles 0..136 never touch the pool boundary (2191 < 2205): no mask
    #pragma unroll
    for (int v = 0; v < 8; ++v) top3_insert(cprev[v], t0, t1, t2);
    cprev = ccur;
  }

  // ---- epilogue: tile 137 with pool-boundary mask
  {
    const int lim = POOL - ((STILES - 1) * 16 + hi * 8);   // valid v < lim
    #pragma unroll
    for (int v = 0; v < 8; ++v) {
      float x = (v < lim) ? cprev[v] : NEG_BIG;
      top3_insert(x, t0, t1, t2);
    }
  }

  // Merge the two disjoint-half top-3 lists for the same query row.
  float ua0 = t0, ua1 = t1, ua2 = t2;
  float ub0 = swap16(t0), ub1 = swap16(t1), ub2 = swap16(t2);
  float s = 0.0f;
  #pragma unroll
  for (int i = 0; i < 3; ++i) {
    bool ta = ua0 >= ub0;
    s += ta ? ua0 : ub0;
    float na0 = ta ? ua1 : ua0, na1 = ta ? ua2 : ua1, na2 = ta ? NEG_BIG : ua2;
    float nb0 = ta ? ub0 : ub1, nb1 = ta ? ub1 : ub2, nb2 = ta ? ub2 : NEG_BIG;
    ua0 = na0; ua1 = na1; ua2 = na2;
    ub0 = nb0; ub1 = nb1; ub2 = nb2;
  }
  if (hi == 0) {
    const int qrow = qtile * 16 + (lane & 15);
    const int b = qrow / HW;                     // image index
    atomicAdd(&acc[b * CLASSES + cls], s);       // sum over query positions
  }
}

// ---------------------------------------------------------------------------
// Kernel 5: clip + geometric mean over av=2  ->  out[16][5]
// ---------------------------------------------------------------------------
__global__ void finalize(const float* __restrict__ acc, float* __restrict__ out) {
  int i = blockIdx.x * blockDim.x + threadIdx.x;
  if (i >= (BQ / 2) * CLASSES) return;
  int e = i / CLASSES, c = i % CLASSES;
  float v0 = fmaxf(acc[(2 * e + 0) * CLASSES + c], EPSV);
  float v1 = fmaxf(acc[(2 * e + 1) * CLASSES + c], EPSV);
  out[i] = sqrtf(v0 * v1);   // exp(mean(log(.))) for av_num == 2
}

// ---------------------------------------------------------------------------
extern "C" void kernel_launch(void* const* d_in, const int* in_sizes, int n_in,
                              void* d_out, int out_size, void* d_ws, size_t ws_size,
                              hipStream_t stream) {
  const float* q = (const float*)d_in[0];
  const float* S = (const float*)d_in[1];
  // av_num (=2) and SAV_num (=5) are compile-time constants here.

  const size_t B_HALFS = (size_t)QTILES * 2 * 32 * 16;           // 903168
  const size_t A_HALFS = (size_t)CLASSES * STILES * 2 * 32 * 16; // 706560

  _Float16* bfrag = (_Float16*)d_ws;
  _Float16* afrag = bfrag + B_HALFS;
  float*    rnq   = (float*)(afrag + A_HALFS);
  float*    rns   = rnq + MQ;
  float*    acc   = rns + NSUP * HW;
  float*    out   = (float*)d_out;

  const int normN = MQ + NSUP * HW;
  norm_init<<<(normN + 255) / 256, 256, 0, stream>>>(q, S, rnq, rns, acc);
  pack_b<<<(QTILES * 2 * 32 + 255) / 256, 256, 0, stream>>>(q, rnq, bfrag);
  pack_a<<<(CLASSES * STILES * 2 * 32 + 255) / 256, 256, 0, stream>>>(S, rns, afrag);
  knn_wmma<<<(QTILES * CLASSES + 3) / 4, 128, 0, stream>>>(afrag, bfrag, acc);
  finalize<<<1, 128, 0, stream>>>(acc, out);
}
